// LSTM_6064493822068
// MI455X (gfx1250) — matrix operand order repaired and checked by reading, not compiled, not run on hardware
//
#include <hip/hip_runtime.h>

// ---------------------------------------------------------------------------
// LSTM on gfx1250 (MI455X), compile-only target.
//
// Fused [x_t | h] @ [W_i ; W_h] per timestep.
//  - Phase 0: swizzle W_i (256x1024) + W_h (256x1024) -> f16 B-fragment
//             layout in workspace (1 MB, L2-resident).
//  - Phase 1: persistent recurrent kernel, 4 workgroups x 512 threads.
//             WG b owns batch rows [16b, 16b+16). Wave w (0..15) owns gate
//             tiles {w, w+16, w+32, w+48} (N-tiles of 16) => h columns
//             [16w, 16w+16). c-state lives in wave registers (v8f), h is
//             exchanged via LDS (f16) with two barriers per step.
//
// Round-1 fix: anti-LICM opacity on the weight access (prevents a
//              512-VGPR hoist + scratch spill of all B tiles).
// Round-2 fix: opacity on an integer offset (not the pointer) so the global
//              address space survives -> global_load_b128, not flat_load.
// Round-3 fix: explicit 1-iteration software pipeline in source.
// Round-4 fix: the machine scheduler flattened the pipeline (waits of 0x0
//              before every WMMA). Now enforce it with sched_group_barrier:
//              per kt: [8 VMEM reads][2 DS reads][4 WMMA] so loads are
//              issued one iteration ahead of their consuming WMMAs.
// ---------------------------------------------------------------------------

typedef _Float16 f16;
typedef __attribute__((ext_vector_type(16))) _Float16 v16h;
typedef __attribute__((ext_vector_type(8)))  _Float16 v8h;
typedef __attribute__((ext_vector_type(8)))  float    v8f;
typedef __attribute__((ext_vector_type(4)))  float    v4f;

#define B_SZ 64
#define T_SZ 2048
#define D_IN 256
#define H_SZ 256
#define KTOT 512           // D_IN + H_SZ
#define NGATE 1024         // 4*H
#define NKT  (KTOT / 32)   // 16 K-tiles of 32
#define NNT  (NGATE / 16)  // 64 N-tiles of 16

// sched_group_barrier masks (LLVM AMDGPU SchedGroupMask)
#define SG_MFMA      0x008   // matrix (WMMA) ops
#define SG_VMEM_READ 0x020
#define SG_DS_READ   0x100

// ---------------------------------------------------------------------------
// Phase 0: convert W_i/W_h (f32 row-major [K,1024]) into f16 tiles laid out
// in the k order this kernel's A loads also use, one contiguous 32B chunk per lane:
//   tile (kt, nt): lane l holds column n = nt*16 + (l&15),
//                  K values k = kt*32 + (l>>4)*16 + e, e = 0..15 (packed f16).
// Storage: wt[(((kt*64 + nt)*32 + l)*16 + e]
// ---------------------------------------------------------------------------
__global__ void lstm_prep_weights(const float* __restrict__ Wi,
                                  const float* __restrict__ Wh,
                                  f16* __restrict__ wt) {
  int idx = blockIdx.x * blockDim.x + threadIdx.x;   // [0, 16*64*32*16)
  int e  = idx & 15;
  int l  = (idx >> 4) & 31;
  int nt = (idx >> 9) & 63;
  int kt = (idx >> 15) & 15;
  int n = nt * 16 + (l & 15);
  int k = kt * 32 + ((l >> 4) * 16 + e);
  float v = (k < D_IN) ? Wi[(size_t)k * NGATE + n]
                       : Wh[(size_t)(k - D_IN) * NGATE + n];
  wt[idx] = (f16)v;
}

// ---------------------------------------------------------------------------
// Phase 1: persistent recurrent kernel. 4 blocks x 512 threads (16 waves).
// ---------------------------------------------------------------------------
__launch_bounds__(512, 1)
__global__ void lstm_recurrent(const float* __restrict__ x,
                               const f16*   __restrict__ wt,
                               const float* __restrict__ bias,
                               float* __restrict__ out) {
  __shared__ f16 lds_x[16 * D_IN];   // x_t tile, f16 row-major [16][256]
  __shared__ f16 lds_h[16 * H_SZ];   // h    tile, f16 row-major [16][256]

  const int tid   = threadIdx.x;
  const int wave  = tid >> 5;        // 0..15
  const int lane  = tid & 31;
  const int lhalf = lane >> 4;       // 0/1  (WMMA half-wave)
  const int l15   = lane & 15;
  const int b0    = blockIdx.x * 16; // batch row base

  // h starts at zero
  for (int i = tid; i < 16 * H_SZ; i += 512) lds_h[i] = (f16)0.f;

  // this lane's h column and per-gate bias (gate order i,f,g,o; each 256 wide)
  const int ncol = wave * 16 + l15;
  const float bi = bias[0 * H_SZ + ncol];
  const float bf = bias[1 * H_SZ + ncol];
  const float bg = bias[2 * H_SZ + ncol];
  const float bo = bias[3 * H_SZ + ncol];

  v8f cstate = {};        // persistent c tile (16x16 f32, D-matrix layout)
  float hlast[8];         // last h values (f32) for h_T output
#pragma unroll
  for (int r = 0; r < 8; ++r) hlast[r] = 0.f;

  float* Hs = out;                                      // [64,2048,256]
  float* hT = out + (size_t)B_SZ * T_SZ * H_SZ;         // [64,256]
  float* cT = hT + (size_t)B_SZ * H_SZ;                 // [64,256]

  // Per-wave/lane byte-layout constants for B tiles:
  //   tile (kt, nt=g*16+wave): halves offset ((kt*64 + nt)*32 + lane)*16.
  const unsigned bofs_lane = (unsigned)lane * 16;

  for (int t = 0; t < T_SZ; ++t) {
    // Opaque zero offset, redefined every iteration: B-tile addresses depend
    // on it, so LLVM cannot hoist the weight loads out of the t-loop (which
    // spilled 512 VGPRs/lane to scratch in round 0). Integer (not pointer)
    // opacity preserves the global address space -> global_load_b128.
    unsigned tofs = 0;
    asm volatile("" : "+s"(tofs));
    const f16* wtl = wt + tofs;

    // B tile fetch for (kt, gate g) of this wave.
#define LOADB(kt_, g_) \
    (*(const v16h*)(wtl + ((unsigned)(((kt_) * NNT) + (g_) * 16 + wave) * 512u) \
                        + bofs_lane))

    // ---- prologue: issue kt=0 B tiles BEFORE the barrier (no LDS dep) ----
    v16h b0v = LOADB(0, 0);
    v16h b1v = LOADB(0, 1);
    v16h b2v = LOADB(0, 2);
    v16h b3v = LOADB(0, 3);

    // ---- cooperative stage x_t -> LDS (f32 -> f16), 8 elements/thread ----
    {
      const int m  = tid >> 5;            // row 0..15
      const int c0 = (tid & 31) * 8;      // col base
      const float* src = x + ((size_t)(b0 + m) * T_SZ + t) * D_IN + c0;
      v4f a0 = *(const v4f*)(src);
      v4f a1 = *(const v4f*)(src + 4);
      if (t + 1 < T_SZ) __builtin_prefetch(src + D_IN, 0, 0);  // next step
      f16* dst = &lds_x[m * D_IN + c0];
#pragma unroll
      for (int j = 0; j < 4; ++j) { dst[j] = (f16)a0[j]; dst[4 + j] = (f16)a1[j]; }
    }
    __syncthreads();   // barrier 1: x_t staged, previous h writes visible

    // A fragment (16x32 f16) base for K-tile kt. This kernel's contiguous layout: lane row
    // M = l&15, K = (lane>=16 ? 16 : 0) + kt*32 + {0..15}.
#define ABASE(kt_) ((kt_) < 8 \
      ? &lds_x[l15 * D_IN + (kt_) * 32 + lhalf * 16] \
      : &lds_h[l15 * H_SZ + ((kt_) - 8) * 32 + lhalf * 16])

    v8h alo = *(const v8h*)(ABASE(0));
    v8h ahi = *(const v8h*)(ABASE(0) + 8);

    // ---- gates = [x_t | h] @ W, K = 512, software-pipelined depth 1 ----
    v8f acc[4] = {v8f{}, v8f{}, v8f{}, v8f{}};
#pragma unroll
    for (int kt = 0; kt < NKT; ++kt) {
      // issue next iteration's loads before consuming current buffers
      v16h n0 = b0v, n1 = b1v, n2 = b2v, n3 = b3v;
      v8h nlo = alo, nhi = ahi;
      if (kt + 1 < NKT) {
        n0 = LOADB(kt + 1, 0);
        n1 = LOADB(kt + 1, 1);
        n2 = LOADB(kt + 1, 2);
        n3 = LOADB(kt + 1, 3);
        nlo = *(const v8h*)(ABASE(kt + 1));
        nhi = *(const v8h*)(ABASE(kt + 1) + 8);
      }
      v16h a = __builtin_shufflevector(alo, ahi,
                 0, 1, 2, 3, 4, 5, 6, 7, 8, 9, 10, 11, 12, 13, 14, 15);
      acc[0] = __builtin_amdgcn_wmma_f32_16x16x32_f16(
          false, a, false, b0v, (short)0, acc[0], false, false);
      acc[1] = __builtin_amdgcn_wmma_f32_16x16x32_f16(
          false, a, false, b1v, (short)0, acc[1], false, false);
      acc[2] = __builtin_amdgcn_wmma_f32_16x16x32_f16(
          false, a, false, b2v, (short)0, acc[2], false, false);
      acc[3] = __builtin_amdgcn_wmma_f32_16x16x32_f16(
          false, a, false, b3v, (short)0, acc[3], false, false);
      b0v = n0; b1v = n1; b2v = n2; b3v = n3;   // SSA rotation, no movs
      alo = nlo; ahi = nhi;

      // Enforce the pipeline on the machine scheduler (IGLP): per iteration,
      // 8 global B loads (kt+1), then 2 LDS A loads (kt+1), then 4 WMMAs (kt).
      // Dependencies then guarantee loads run one iteration ahead.
      if (kt + 1 < NKT) {
        __builtin_amdgcn_sched_group_barrier(SG_VMEM_READ, 8, 0);
        __builtin_amdgcn_sched_group_barrier(SG_DS_READ,   2, 0);
      }
      __builtin_amdgcn_sched_group_barrier(SG_MFMA, 4, 0);
    }
#undef LOADB
#undef ABASE
    __syncthreads();   // barrier 2: all waves done reading lds_h / lds_x

    // ---- LSTM cell (per lane: 8 rows of this wave's 16x16 tile) ----
#pragma unroll
    for (int r = 0; r < 8; ++r) {
      float iv = acc[0][r] + bi;
      float fv = acc[1][r] + bf;
      float gv = acc[2][r] + bg;
      float ov = acc[3][r] + bo;
      float si = 1.f / (1.f + __expf(-iv));
      float sf = 1.f / (1.f + __expf(-fv));
      float so = 1.f / (1.f + __expf(-ov));
      float e2g = __expf(2.f * gv);
      float tg  = (e2g - 1.f) / (e2g + 1.f);     // tanh(g)
      float cn  = sf * cstate[r] + si * tg;
      float e2c = __expf(2.f * cn);
      float th  = (e2c - 1.f) / (e2c + 1.f);     // tanh(c)
      float hn  = so * th;
      cstate[r] = cn;
      hlast[r]  = hn;
      const int m = r + lhalf * 8;               // D-matrix row mapping
      lds_h[m * H_SZ + ncol] = (f16)hn;          // publish h for step t+1
      Hs[((size_t)(b0 + m) * T_SZ + t) * H_SZ + ncol] = hn;
    }
  }

  // ---- final h_T, c_T ----
#pragma unroll
  for (int r = 0; r < 8; ++r) {
    const int m = r + lhalf * 8;
    hT[(size_t)(b0 + m) * H_SZ + ncol] = hlast[r];
    cT[(size_t)(b0 + m) * H_SZ + ncol] = cstate[r];
  }
}

// ---------------------------------------------------------------------------
extern "C" void kernel_launch(void* const* d_in, const int* in_sizes, int n_in,
                              void* d_out, int out_size, void* d_ws, size_t ws_size,
                              hipStream_t stream) {
  const float* x    = (const float*)d_in[0];   // [64, 2048, 256]
  const float* Wi   = (const float*)d_in[1];   // [256, 1024]
  const float* Wh   = (const float*)d_in[2];   // [256, 1024]
  const float* bias = (const float*)d_in[3];   // [1024]
  f16* wt = (f16*)d_ws;                        // 512*1024 f16 = 1 MB

  // Phase 0: weight swizzle -> f16 B-fragment layout (L2-resident, 1 MB)
  lstm_prep_weights<<<2048, 256, 0, stream>>>(Wi, Wh, wt);

  // Phase 1: recurrence, 4 batch-slices x 16 waves
  lstm_recurrent<<<4, 512, 0, stream>>>(x, wt, bias, (float*)d_out);
}
